// BiMambaBlock_15023795601428
// MI455X (gfx1250) — matrix operand order repaired and checked by reading, not compiled, not run on hardware
//
#include <hip/hip_runtime.h>
#include <hip/hip_bf16.h>

// ---------------------------------------------------------------------------
// BiMamba block for MI455X (gfx1250, wave32, WMMA, chunked parallel scan).
// ---------------------------------------------------------------------------

#define D_MODEL 128
#define D_STATE 32
#define D_CONV  8
#define D_INNER 256
#define DT_RANK 8
#define XDBL_LD 80    // 72 real columns, ld padded to 80
#define XPROJ_PAD 128 // x_proj rows padded 72 -> 128 (two 64-wide wave tiles)
#define BB 4
#define LL 2048
#define BL (BB * LL)
#define NCHUNK 32     // scan chunks
#define CLEN  (LL / NCHUNK)   // 64 tokens per chunk

typedef __attribute__((ext_vector_type(16))) __bf16 v16bf;
typedef __attribute__((ext_vector_type(8)))  __bf16 v8bf;
typedef __attribute__((ext_vector_type(8)))  float  v8f;

// ---- WMMA fragment loaders (bf16, 16x16x32; contiguous k = 16*(lane/16) + s, same for A and B) --

// A-matrix 16x32 (MxK): lane<16 -> row M=lane, K = k0..k0+15
//                       lane>=16 -> row M=lane-16, K = k0+16..k0+31
__device__ __forceinline__ v16bf load_a_frag(const __bf16* __restrict__ base,
                                             int ld, int row0, int k0, int lane) {
  const __bf16* p = base + (size_t)(row0 + (lane & 15)) * ld + k0 + (lane & 16);
  v8bf lo = *reinterpret_cast<const v8bf*>(p);
  v8bf hi = *reinterpret_cast<const v8bf*>(p + 8);
  v16bf r;
#pragma unroll
  for (int j = 0; j < 8; ++j) { r[j] = lo[j]; r[8 + j] = hi[j]; }
  return r;
}

// B-matrix 32x16 (KxN), sourced from W stored row-major (N,K):
// lane<16 -> col N=lane, K = k0..k0+15 ; lane>=16 -> col N=lane-16, K = k0+16..k0+31
__device__ __forceinline__ v16bf load_b_frag(const __bf16* __restrict__ base,
                                             int ld, int col0, int k0, int lane) {
  const __bf16* p = base + (size_t)(col0 + (lane & 15)) * ld + k0 + (lane & 16);
  return *reinterpret_cast<const v16bf*>(p);
}

// ---- WMMA GEMM: C[m,n] (+)= sum_k A[m,k] * W[n,k] --------------------------
// A: (M,K) bf16 row-major, W: (Npad,K) bf16 row-major (zero-padded),
// C: f32 leading-dim ldc, only columns < Nstore are written.
// One wave computes a 16x64 strip: A fragment loaded once, 4 WMMAs per k-step.
__global__ void gemm_bf16_wmma(const __bf16* __restrict__ A,
                               const __bf16* __restrict__ W,
                               float* __restrict__ C,
                               int Npad, int Nstore, int K, int ldc, int accumulate) {
  int lane = threadIdx.x & 31;
  int wave = threadIdx.x >> 5;
  int m0 = blockIdx.x * 16;
  int nbase = (blockIdx.y * 8 + wave) * 64;
  if (nbase >= Npad) return;             // wave-uniform exit (EXEC stays all-1s)

  int crow = m0 + ((lane >> 4) << 3);
  int ccl  = lane & 15;

  v8f acc[4];
#pragma unroll
  for (int tn = 0; tn < 4; ++tn) {
    int n0 = nbase + tn * 16;
    if (accumulate && n0 < Nstore) {
#pragma unroll
      for (int r = 0; r < 8; ++r) acc[tn][r] = C[(size_t)(crow + r) * ldc + n0 + ccl];
    } else {
#pragma unroll
      for (int r = 0; r < 8; ++r) acc[tn][r] = 0.0f;
    }
  }

  for (int k0 = 0; k0 < K; k0 += 32) {
    v16bf a = load_a_frag(A, K, m0, k0, lane);
#pragma unroll
    for (int tn = 0; tn < 4; ++tn) {
      v16bf b = load_b_frag(W, K, nbase + tn * 16, k0, lane);
      acc[tn] = __builtin_amdgcn_wmma_f32_16x16x32_bf16(
          false, a, false, b, (short)0, acc[tn], false, false);
    }
  }

#pragma unroll
  for (int tn = 0; tn < 4; ++tn) {
    int n0 = nbase + tn * 16;
    if (n0 < Nstore) {   // wave-uniform
#pragma unroll
      for (int r = 0; r < 8; ++r) C[(size_t)(crow + r) * ldc + n0 + ccl] = acc[tn][r];
    }
  }
}

// ---- LayerNorm over D_MODEL, emits bf16 ------------------------------------
__global__ void layernorm_bf16(const float* __restrict__ x,
                               const float* __restrict__ g,
                               const float* __restrict__ bta,
                               __bf16* __restrict__ xn) {
  int t = blockIdx.x;
  int tid = threadIdx.x;           // 0..127
  __shared__ float red[D_MODEL];
  float v = x[(size_t)t * D_MODEL + tid];
  red[tid] = v;
  __syncthreads();
  for (int s = 64; s > 0; s >>= 1) { if (tid < s) red[tid] += red[tid + s]; __syncthreads(); }
  float mu = red[0] * (1.0f / D_MODEL);
  __syncthreads();
  float d = v - mu;
  red[tid] = d * d;
  __syncthreads();
  for (int s = 64; s > 0; s >>= 1) { if (tid < s) red[tid] += red[tid + s]; __syncthreads(); }
  float var = red[0] * (1.0f / D_MODEL);
  float xnv = d * rsqrtf(var + 1e-5f) * g[tid] + bta[tid];
  xn[(size_t)t * D_MODEL + tid] = (__bf16)xnv;
}

// ---- weight conversion helpers ---------------------------------------------
__global__ void cvt_bf16(const float* __restrict__ s, __bf16* __restrict__ d, int n) {
  int i = blockIdx.x * blockDim.x + threadIdx.x;
  if (i < n) d[i] = (__bf16)s[i];
}
__global__ void cvt_pad_rows(const float* __restrict__ s, __bf16* __restrict__ d,
                             int rows_src, int rows_dst, int cols) {
  int i = blockIdx.x * blockDim.x + threadIdx.x;
  if (i >= rows_dst * cols) return;
  int r = i / cols;
  d[i] = (r < rows_src) ? (__bf16)s[i] : (__bf16)0.0f;
}
// W_comb[n,k] = sum_j gate_w[n, dir*128 + j] * out_proj[j, k]   (128 x 256)
__global__ void combine_w(const float* __restrict__ gate_w,
                          const float* __restrict__ out_proj,
                          __bf16* __restrict__ wcomb, int dir) {
  int idx = blockIdx.x * blockDim.x + threadIdx.x;   // 0..32767
  int n = idx >> 8, k = idx & 255;
  float acc = 0.0f;
#pragma unroll 4
  for (int j = 0; j < D_MODEL; ++j)
    acc += gate_w[n * (2 * D_MODEL) + dir * D_MODEL + j] * out_proj[j * D_INNER + k];
  wcomb[idx] = (__bf16)acc;
}

// ---- depthwise causal conv (direction-aware) + SiLU ------------------------
__global__ void conv_silu(const float* __restrict__ xz,
                          const float* __restrict__ cw,
                          const float* __restrict__ cb,
                          float* __restrict__ u,
                          __bf16* __restrict__ ubf, int dir) {
  int bt = blockIdx.x;
  int d = threadIdx.x;
  int b = bt / LL, t = bt % LL;
  float acc = cb[d];
#pragma unroll
  for (int j = 0; j < D_CONV; ++j) {
    int tt = (dir == 0) ? (t - (D_CONV - 1) + j) : (t + (D_CONV - 1) - j);
    float xv = (tt >= 0 && tt < LL) ? xz[((size_t)b * LL + tt) * 512 + d] : 0.0f;
    acc = fmaf(xv, cw[d * D_CONV + j], acc);
  }
  float s = acc / (1.0f + __expf(-acc));
  u[(size_t)bt * D_INNER + d] = s;
  ubf[(size_t)bt * D_INNER + d] = (__bf16)s;
}

// ---- delta = softplus(dt @ dt_projT + dt_bias) -----------------------------
__global__ void delta_softplus(const float* __restrict__ xdbl,
                               const float* __restrict__ dtw,
                               const float* __restrict__ dtb,
                               float* __restrict__ delta) {
  int bt = blockIdx.x;
  int d = threadIdx.x;
  const float* dt = xdbl + (size_t)bt * XDBL_LD;
  float acc = dtb[d];
#pragma unroll
  for (int k = 0; k < DT_RANK; ++k) acc = fmaf(dtw[d * DT_RANK + k], dt[k], acc);
  float sp = (acc > 20.0f) ? acc : __logf(1.0f + __expf(acc));
  delta[(size_t)bt * D_INNER + d] = sp;
}

// ---- chunked scan, pass 1: per-chunk decay product + local partial state ---
// grid (NCHUNK, B); 256 threads = one channel each.
__global__ void scan_partial(const float* __restrict__ delta,
                             const float* __restrict__ u,
                             const float* __restrict__ xdbl,
                             const float* __restrict__ A_log,
                             float* __restrict__ Pdec,
                             float* __restrict__ hfin, int dir) {
  int c = blockIdx.x, b = blockIdx.y, d = threadIdx.x;
  __shared__ float sB[CLEN][D_STATE];   // 8 KB
  for (int i = threadIdx.x; i < CLEN * D_STATE; i += 256) {
    int s = i >> 5, e = i & 31;
    int p = c * CLEN + s;
    int t = dir ? (LL - 1 - p) : p;
    sB[s][e] = xdbl[((size_t)b * LL + t) * XDBL_LD + DT_RANK + e];
  }
  {  // warm L2/WGP$ for the streamed per-channel inputs
    int p0 = c * CLEN;
    int t0 = dir ? (LL - 1 - p0) : p0;
    size_t base0 = ((size_t)b * LL + t0) * D_INNER + d;
    __builtin_prefetch(delta + base0, 0, 0);
    __builtin_prefetch(u + base0, 0, 0);
  }
  __syncthreads();

  float A[D_STATE], P[D_STATE], h[D_STATE];
#pragma unroll
  for (int n = 0; n < D_STATE; ++n) {
    A[n] = -__expf(A_log[d * D_STATE + n]);
    P[n] = 1.0f;
    h[n] = 0.0f;
  }
  for (int s = 0; s < CLEN; ++s) {
    int p = c * CLEN + s;
    int t = dir ? (LL - 1 - p) : p;
    size_t base = (size_t)b * LL + t;
    float dl = delta[base * D_INNER + d];
    float uv = u[base * D_INNER + d];
    float du = dl * uv;
#pragma unroll
    for (int n = 0; n < D_STATE; ++n) {
      float dA = __expf(dl * A[n]);
      P[n] *= dA;
      h[n] = fmaf(dA, h[n], du * sB[s][n]);
    }
  }
  size_t ob = (((size_t)b * NCHUNK + c) * D_INNER + d) * D_STATE;
#pragma unroll
  for (int n = 0; n < D_STATE; ++n) { Pdec[ob + n] = P[n]; hfin[ob + n] = h[n]; }
}

// ---- chunked scan, pass 2: serial combine over 32 chunk summaries ----------
__global__ void scan_combine(const float* __restrict__ Pdec,
                             const float* __restrict__ hfin,
                             float* __restrict__ h0a) {
  int b = blockIdx.x, d = threadIdx.x;
  float h0[D_STATE];
#pragma unroll
  for (int n = 0; n < D_STATE; ++n) h0[n] = 0.0f;
  for (int c = 0; c < NCHUNK; ++c) {
    size_t base = (((size_t)b * NCHUNK + c) * D_INNER + d) * D_STATE;
#pragma unroll
    for (int n = 0; n < D_STATE; ++n) h0a[base + n] = h0[n];
#pragma unroll
    for (int n = 0; n < D_STATE; ++n) h0[n] = fmaf(Pdec[base + n], h0[n], hfin[base + n]);
  }
}

// ---- chunked scan, pass 3: replay chunk from h0, fused D-skip + gate -------
__global__ void scan_final(const float* __restrict__ delta,
                           const float* __restrict__ u,
                           const float* __restrict__ xdbl,
                           const float* __restrict__ xz,
                           const float* __restrict__ A_log,
                           const float* __restrict__ Dv,
                           const float* __restrict__ h0a,
                           __bf16* __restrict__ ybf, int dir) {
  int c = blockIdx.x, b = blockIdx.y, d = threadIdx.x;
  __shared__ float sBC[CLEN][2 * D_STATE];   // 16 KB: [0:32]=B, [32:64]=C
  for (int i = threadIdx.x; i < CLEN * 2 * D_STATE; i += 256) {
    int s = i >> 6, e = i & 63;
    int p = c * CLEN + s;
    int t = dir ? (LL - 1 - p) : p;
    sBC[s][e] = xdbl[((size_t)b * LL + t) * XDBL_LD + DT_RANK + e];
  }
  {
    int p0 = c * CLEN;
    int t0 = dir ? (LL - 1 - p0) : p0;
    size_t base0 = ((size_t)b * LL + t0) * D_INNER + d;
    __builtin_prefetch(delta + base0, 0, 0);
    __builtin_prefetch(u + base0, 0, 0);
  }
  __syncthreads();

  float A[D_STATE], h[D_STATE];
  size_t ob = (((size_t)b * NCHUNK + c) * D_INNER + d) * D_STATE;
#pragma unroll
  for (int n = 0; n < D_STATE; ++n) {
    A[n] = -__expf(A_log[d * D_STATE + n]);
    h[n] = h0a[ob + n];
  }
  float Dd = Dv[d];

  for (int s = 0; s < CLEN; ++s) {
    int p = c * CLEN + s;
    int t = dir ? (LL - 1 - p) : p;
    size_t base = (size_t)b * LL + t;
    float dl = delta[base * D_INNER + d];
    float uv = u[base * D_INNER + d];
    float du = dl * uv;
    float y = 0.0f;
#pragma unroll
    for (int n = 0; n < D_STATE; ++n) {
      float dA = __expf(dl * A[n]);
      h[n] = fmaf(dA, h[n], du * sBC[s][n]);
      y = fmaf(h[n], sBC[s][32 + n], y);
    }
    float zv = xz[base * 512 + D_INNER + d];
    float yg = (y + uv * Dd) * (zv / (1.0f + __expf(-zv)));
    ybf[base * D_INNER + d] = (__bf16)yg;
  }
}

// ---- out init: out = x + gate_b --------------------------------------------
__global__ void init_out(const float* __restrict__ x,
                         const float* __restrict__ gb,
                         float* __restrict__ out, int total) {
  int i = blockIdx.x * blockDim.x + threadIdx.x;
  if (i < total) out[i] = x[i] + gb[i & (D_MODEL - 1)];
}

// ---------------------------------------------------------------------------
extern "C" void kernel_launch(void* const* d_in, const int* in_sizes, int n_in,
                              void* d_out, int out_size, void* d_ws, size_t ws_size,
                              hipStream_t stream) {
  (void)in_sizes; (void)n_in; (void)out_size; (void)ws_size;

  const float* x      = (const float*)d_in[0];
  const float* norm_g = (const float*)d_in[1];
  const float* norm_b = (const float*)d_in[2];
  const float* gate_w = (const float*)d_in[21];
  const float* gate_b = (const float*)d_in[22];
  auto P = [&](int dir, int which) -> const float* {
    return (const float*)d_in[3 + dir * 9 + which];
  };
  // which: 0=in_proj 1=conv_w 2=conv_b 3=x_proj 4=dt_proj 5=dt_bias 6=A_log 7=D 8=out_proj

  float* out = (float*)d_out;

  char* ws = (char*)d_ws;
  size_t off = 0;
  auto alloc = [&](size_t bytes) -> char* {
    char* p = ws + off;
    off = (off + bytes + 255) & ~(size_t)255;
    return p;
  };
  __bf16* xn_bf = (__bf16*)alloc((size_t)BL * D_MODEL * 2);
  __bf16 *inproj_bf[2], *xproj_bf[2], *wcomb_bf[2], *ubf[2], *ybf[2];
  float  *xz[2], *uf[2], *xdbl[2], *dl[2], *Pdec[2], *hfin[2], *h0a[2];
  for (int dir = 0; dir < 2; ++dir) {
    inproj_bf[dir] = (__bf16*)alloc((size_t)512 * D_MODEL * 2);
    xproj_bf[dir]  = (__bf16*)alloc((size_t)XPROJ_PAD * D_INNER * 2);
    wcomb_bf[dir]  = (__bf16*)alloc((size_t)D_MODEL * D_INNER * 2);
    xz[dir]        = (float*)alloc((size_t)BL * 512 * 4);
    uf[dir]        = (float*)alloc((size_t)BL * D_INNER * 4);
    ubf[dir]       = (__bf16*)alloc((size_t)BL * D_INNER * 2);
    xdbl[dir]      = (float*)alloc((size_t)BL * XDBL_LD * 4);
    dl[dir]        = (float*)alloc((size_t)BL * D_INNER * 4);
    ybf[dir]       = (__bf16*)alloc((size_t)BL * D_INNER * 2);
    Pdec[dir]      = (float*)alloc((size_t)BB * NCHUNK * D_INNER * D_STATE * 4);
    hfin[dir]      = (float*)alloc((size_t)BB * NCHUNK * D_INNER * D_STATE * 4);
    h0a[dir]       = (float*)alloc((size_t)BB * NCHUNK * D_INNER * D_STATE * 4);
  }

  // ---- stage 0: layernorm + weight prep ----
  layernorm_bf16<<<BL, D_MODEL, 0, stream>>>(x, norm_g, norm_b, xn_bf);
  for (int dir = 0; dir < 2; ++dir) {
    cvt_bf16<<<(512 * D_MODEL + 255) / 256, 256, 0, stream>>>(P(dir, 0), inproj_bf[dir],
                                                              512 * D_MODEL);
    cvt_pad_rows<<<(XPROJ_PAD * D_INNER + 255) / 256, 256, 0, stream>>>(
        P(dir, 3), xproj_bf[dir], 72, XPROJ_PAD, D_INNER);
    combine_w<<<(D_MODEL * D_INNER + 255) / 256, 256, 0, stream>>>(gate_w, P(dir, 8),
                                                                   wcomb_bf[dir], dir);
  }

  // ---- per-direction pipeline ----
  for (int dir = 0; dir < 2; ++dir) {
    // xz = xn @ in_proj^T : M=8192, Npad=512, K=128
    gemm_bf16_wmma<<<dim3(BL / 16, 1), 256, 0, stream>>>(
        xn_bf, inproj_bf[dir], xz[dir], 512, 512, D_MODEL, 512, 0);
    // depthwise conv + SiLU -> u
    conv_silu<<<BL, D_INNER, 0, stream>>>(xz[dir], P(dir, 1), P(dir, 2), uf[dir], ubf[dir], dir);
    // x_dbl = u @ x_proj^T : Npad=128 (zero rows 72..127), Nstore=80, K=256
    gemm_bf16_wmma<<<dim3(BL / 16, 1), 256, 0, stream>>>(
        ubf[dir], xproj_bf[dir], xdbl[dir], XPROJ_PAD, XDBL_LD, D_INNER, XDBL_LD, 0);
    // delta = softplus(dt @ dt_proj^T + dt_bias)
    delta_softplus<<<BL, D_INNER, 0, stream>>>(xdbl[dir], P(dir, 4), P(dir, 5), dl[dir]);
    // chunked selective scan (3 passes)
    scan_partial<<<dim3(NCHUNK, BB), D_INNER, 0, stream>>>(
        dl[dir], uf[dir], xdbl[dir], P(dir, 6), Pdec[dir], hfin[dir], dir);
    scan_combine<<<BB, D_INNER, 0, stream>>>(Pdec[dir], hfin[dir], h0a[dir]);
    scan_final<<<dim3(NCHUNK, BB), D_INNER, 0, stream>>>(
        dl[dir], uf[dir], xdbl[dir], xz[dir], P(dir, 6), P(dir, 7), h0a[dir], ybf[dir], dir);
  }

  // ---- out = x + gate_b + y_f @ Wf^T + y_b @ Wb^T ----
  init_out<<<(BL * D_MODEL + 255) / 256, 256, 0, stream>>>(x, gate_b, out, BL * D_MODEL);
  for (int dir = 0; dir < 2; ++dir) {
    gemm_bf16_wmma<<<dim3(BL / 16, 1), 256, 0, stream>>>(
        ybf[dir], wcomb_bf[dir], out, D_MODEL, D_MODEL, D_INNER, D_MODEL, 1);
  }
}
